// TransformerPosterior_86397562127156
// MI455X (gfx1250) — matrix operand order repaired and checked by reading, not compiled, not run on hardware
//
#include <hip/hip_runtime.h>

typedef __attribute__((ext_vector_type(16))) __bf16 v16bf;
typedef __attribute__((ext_vector_type(8)))  __bf16 v8bf;
typedef __attribute__((ext_vector_type(8)))  float  v8f;
typedef __attribute__((ext_vector_type(4)))  int    v4i;

#define NEGV (-1e9f)

// Problem constants (fixed by the reference)
constexpr int Bc = 32, Tc = 2000, Sc = 400, NMc = 80, Hc = 256, Kc = 5;
constexpr int NZ = 8;                      // T-chunks for softmax stats

// gfx1250 async load-to-LDS path (guarded; falls back to sync copy)
#if defined(__has_builtin)
#if __has_builtin(__builtin_amdgcn_global_load_async_to_lds_b128) && \
    __has_builtin(__builtin_amdgcn_s_wait_asynccnt)
#define USE_ASYNC_LDS 1
#endif
#endif
#ifndef USE_ASYNC_LDS
#define USE_ASYNC_LDS 0
#endif

// ---- WMMA fragment maps (wave32, CDNA5 ISA 7.12.2) ----
// A (16x32 bf16, MxK): row M = lane%16; elem e -> K = (lane/16)*16 + e  (contiguous order used here, same as B)
// B (32x16 bf16, KxN): col N = lane%16; elem e -> K = (lane/16)*16 + e  (contiguous order chosen here, not the ISA table order)
// C/D (16x16 f32): acc reg r -> row M = r + 8*(lane/16), col N = lane%16

// ---------------- f32 -> bf16 conversion (grid-stride) ----------------
__global__ void cvt_f32_bf16(const float* __restrict__ src, __bf16* __restrict__ dst,
                             long long n) {
  long long i = (long long)blockIdx.x * blockDim.x + threadIdx.x;
  long long stride = (long long)gridDim.x * blockDim.x;
  for (; i < n; i += stride) dst[i] = (__bf16)src[i];
}

// ------------- weight f32[R,Cout] -> bf16 transposed [Cout,R] ---------
__global__ void cvt_w_transpose(const float* __restrict__ src, __bf16* __restrict__ dst,
                                int R, int Cout) {
  int idx = blockIdx.x * blockDim.x + threadIdx.x;
  if (idx >= R * Cout) return;
  int r = idx / Cout, co = idx - r * Cout;
  dst[(size_t)co * R + r] = (__bf16)src[idx];
}

// ---------------- Conv1d (K=5, SAME) + bias + tanh via WMMA bf16 ------
// EDGE=false: all 5 taps guaranteed in-range (no selects, pure vector loads)
template <int CIN, bool EDGE>
__device__ __forceinline__ void conv_accum(const __bf16* __restrict__ xb,
                                           const __bf16* __restrict__ wcol,
                                           int trow, int T, int asel, int bsel,
                                           v8f& acc) {
  const __bf16 zero = (__bf16)0.0f;
#pragma unroll
  for (int k = 0; k < Kc; ++k) {
    const int tt = trow + k - 2;                  // SAME padding halo
    bool rok = true;
    const __bf16* xrow;
    if (EDGE) {
      rok  = (tt >= 0) && (tt < T);
      xrow = xb + (size_t)(rok ? tt : 0) * CIN;
    } else {
      xrow = xb + (size_t)tt * CIN;
    }
#pragma unroll
    for (int ci0 = 0; ci0 < CIN; ci0 += 32) {
      const int cl = ci0 + bsel;                  // elems 0..7  -> ci = cl + e
      const int ch = ci0 + bsel + 8;              // elems 8..15 -> ci = ch + e
      v8bf lo = {}, hi = {};
      if (cl + 8 <= CIN) lo = *reinterpret_cast<const v8bf*>(xrow + cl);
      if (ch + 8 <= CIN) hi = *reinterpret_cast<const v8bf*>(xrow + ch);
      v16bf a;
#pragma unroll
      for (int e = 0; e < 8; ++e) {
        a[e]     = (EDGE && !rok) ? zero : lo[e];
        a[8 + e] = (EDGE && !rok) ? zero : hi[e];
      }
      v16bf bm = {};
      if (ci0 + bsel + 16 <= CIN)
        bm = *reinterpret_cast<const v16bf*>(wcol + k * CIN + ci0 + bsel);
      acc = __builtin_amdgcn_wmma_f32_16x16x32_bf16(false, a, false, bm,
                                                    (short)0, acc, false, false);
    }
  }
}

// grid: (T/16, Cout/64, B), block 128; wt_t: bf16 weights transposed [Cout, K*CIN]
template <int CIN>
__global__ void conv_tanh_wmma(const __bf16* __restrict__ xin,
                               const __bf16* __restrict__ wt_t,
                               const float* __restrict__ bias,
                               __bf16* __restrict__ out_bf,
                               float* __restrict__ out_f32,
                               int T, int Cout) {
  constexpr int R = Kc * CIN;
  const int b    = blockIdx.z;
  const int t0   = blockIdx.x * 16;
  const int wave = threadIdx.x >> 5;
  const int lane = threadIdx.x & 31;
  const int co0  = blockIdx.y * 64 + wave * 16;
  const int row  = lane & 15;
  const int trow = t0 + row;
  const int col  = co0 + row;
  const int asel = (lane >> 4) << 3;   // 0 or 8 (halves)
  const int bsel = (lane >> 4) << 4;   // 0 or 16 (halves)

  const __bf16* xb   = xin + (size_t)b * T * CIN;
  const __bf16* wcol = wt_t + (size_t)col * R;
  __builtin_prefetch(wcol, 0, 1);      // global_prefetch of this lane's weight column

  v8f acc = {};
  const bool interior = (t0 >= 2) && (t0 + 18 <= T);   // block-uniform
  if (interior) conv_accum<CIN, false>(xb, wcol, trow, T, asel, bsel, acc);
  else          conv_accum<CIN, true >(xb, wcol, trow, T, asel, bsel, acc);

  const float bv = bias[col];
#pragma unroll
  for (int rr = 0; rr < 8; ++rr) {
    int m = rr + ((lane >> 4) << 3);
    int t = t0 + m;
    float v = tanhf(acc[rr] + bv);
    size_t oidx = (size_t)b * T * Cout + (size_t)t * Cout + col;
    out_bf[oidx] = (__bf16)v;
    if (out_f32) out_f32[oidx] = v;
  }
}

// ---------------- W[b,t,s] = sum_h wspec[b,t,h]*src[b,s,h] (WMMA) -----
// grid: (T/16, ceil(Stiles/4), B), block 128. Async-stages 64 src rows to LDS.
__global__ void einsum_wmma(const __bf16* __restrict__ wspec,
                            const __bf16* __restrict__ src,
                            float* __restrict__ Wout) {
  constexpr int T = Tc, S = Sc, H = Hc;
  __shared__ __align__(16) __bf16 lds_src[64 * H];

  const int b    = blockIdx.z;
  const int t0   = blockIdx.x * 16;
  const int wave = threadIdx.x >> 5;
  const int lane = threadIdx.x & 31;
  const int s0b  = blockIdx.y * 64;

  {
    const int nrows = (S - s0b) < 64 ? (S - s0b) : 64;
    const __bf16* gsrc = src + (size_t)b * S * H + (size_t)s0b * H;
    const int nchunks = nrows * (H / 8);          // 16B chunks
    for (int c = threadIdx.x; c < nchunks; c += blockDim.x) {
#if USE_ASYNC_LDS
      __builtin_amdgcn_global_load_async_to_lds_b128(
          (__attribute__((address_space(1))) v4i*)(gsrc + c * 8),
          (__attribute__((address_space(3))) v4i*)(&lds_src[c * 8]),
          0, 0);
#else
      *reinterpret_cast<v8bf*>(&lds_src[c * 8]) =
          *reinterpret_cast<const v8bf*>(gsrc + c * 8);
#endif
    }
#if USE_ASYNC_LDS
    __builtin_amdgcn_s_wait_asynccnt(0);
#endif
    __syncthreads();
  }

  const int stile = blockIdx.y * 4 + wave;
  const int s0    = stile * 16;
  if (s0 >= S) return;                  // wave-uniform guard (after the barrier)

  const int col  = s0 + (lane & 15);
  const int trow = t0 + (lane & 15);
  const __bf16* wsb  = wspec + (size_t)b * T * H + (size_t)trow * H;
  const __bf16* lrow = &lds_src[(size_t)(col - s0b) * H];
  const int asel = (lane >> 4) << 3;
  const int bsel = (lane >> 4) << 4;

  v8f acc = {};
#pragma unroll
  for (int h = 0; h < H; h += 32) {
    v8bf lo = *reinterpret_cast<const v8bf*>(wsb + h + bsel);
    v8bf hi = *reinterpret_cast<const v8bf*>(wsb + h + bsel + 8);
    v16bf a;
#pragma unroll
    for (int e = 0; e < 8; ++e) { a[e] = lo[e]; a[8 + e] = hi[e]; }
    v16bf bm = *reinterpret_cast<const v16bf*>(lrow + h + bsel);   // ds_load
    acc = __builtin_amdgcn_wmma_f32_16x16x32_bf16(false, a, false, bm,
                                                  (short)0, acc, false, false);
  }
#pragma unroll
  for (int rr = 0; rr < 8; ++rr) {
    int t = t0 + rr + ((lane >> 4) << 3);
    Wout[(size_t)b * T * S + (size_t)t * S + col] = acc[rr];
  }
}

// ------------- softmax over T, stage 1: per-T-chunk column stats ------
// grid: (ceil(S/256), B, NZ), block 256
__global__ void colstats_partial(const float* __restrict__ Wg,
                                 float* __restrict__ part, int T, int S, int B) {
  const int b = blockIdx.y;
  const int s = blockIdx.x * blockDim.x + threadIdx.x;
  const int z = blockIdx.z;
  if (s >= S) return;
  const float* base = Wg + (size_t)b * T * S + s;
  const int tb = z * (T / NZ), te = tb + T / NZ;
  float m = -3.4e38f, d = 0.f;
  for (int t = tb; t < te; ++t) {
    float v = base[(size_t)t * S];
    float nm = fmaxf(m, v);
    d = d * __expf(m - nm) + __expf(v - nm);
    m = nm;
  }
  size_t pc = ((size_t)z * B + b) * S + s;
  part[2 * pc]     = m;
  part[2 * pc + 1] = d;
}

// ------------- softmax stage 2: logsumexp-merge the NZ partials -------
__global__ void colstats_merge(const float* __restrict__ part,
                               float* __restrict__ stats, int S, int B) {
  int c = blockIdx.x * blockDim.x + threadIdx.x;
  if (c >= B * S) return;
  float m = -3.4e38f;
#pragma unroll
  for (int z = 0; z < NZ; ++z) m = fmaxf(m, part[2 * ((size_t)z * B * S + c)]);
  float d = 0.f;
#pragma unroll
  for (int z = 0; z < NZ; ++z) {
    size_t p = 2 * ((size_t)z * B * S + c);
    d += part[p + 1] * __expf(part[p] - m);
  }
  stats[2 * c]     = m;
  stats[2 * c + 1] = 1.0f / d;
}

// ------------- softmax stage 3: normalize in place (no divisions) -----
// grid: (ceil(S/256), B, NZ), block 256; each thread walks one column chunk
__global__ void softmax_norm(float* __restrict__ A,
                             const float* __restrict__ stats, int T, int S, int B) {
  const int b = blockIdx.y;
  const int s = blockIdx.x * blockDim.x + threadIdx.x;
  const int z = blockIdx.z;
  if (s >= S) return;
  const int c = b * S + s;
  const float m = stats[2 * c], inv = stats[2 * c + 1];
  float* base = A + (size_t)b * T * S + s;
  const int tb = z * (T / NZ), te = tb + T / NZ;
  for (int t = tb; t < te; ++t)
    base[(size_t)t * S] = __expf(base[(size_t)t * S] - m) * inv;
}

// ---------------- Bayesian DP scan: mu only (serial T) ----------------
// grid: (B), block 512; alpha == 1. pi is produced by the parallel dp_pi.
__global__ void dp_mu_scan(const float* __restrict__ align,
                           const float* __restrict__ mask,
                           float* __restrict__ mu, int T, int S) {
  __shared__ float mp[512];
  const int b = blockIdx.x;
  const int s = threadIdx.x;
  const size_t base = (size_t)b * T * S;

  if (s < S) {
    size_t idx = base + s;
    float m = align[idx] * mask[idx] + (s == 0 ? 0.f : NEGV);
    mu[idx] = m;
    mp[s] = m;
  }
  __syncthreads();

  for (int t = 1; t < T; ++t) {
    float a = 0.f, sh = NEGV;
    if (s < S) {
      a  = mp[s];
      sh = (s > 0) ? mp[s - 1] : NEGV;
    }
    __syncthreads();
    if (s < S) {
      size_t idx = base + (size_t)t * S + s;
      float mx  = fmaxf(a, sh);
      float lse = mx + __logf(__expf(a - mx) + __expf(sh - mx));
      float m   = align[idx] * mask[idx] + lse;
      mu[idx] = m;
      mp[s] = m;
    }
    __syncthreads();
  }
}

// ---------------- pi from stored mu: fully parallel -------------------
// grid: (T, B), block 512; pi written as coalesced float2
__global__ void dp_pi(const float* __restrict__ mu, float2* __restrict__ pi,
                      int T, int S) {
  const int t = blockIdx.x, b = blockIdx.y, s = threadIdx.x;
  if (s >= S) return;
  const size_t rowp = (size_t)b * T * S + (size_t)t * S;
  float2 out;
  if (t == 0) {
    out.x = 1.f; out.y = 0.f;
  } else {
    const float* prev = mu + rowp - S;
    float a  = prev[s];
    float sh = (s > 0) ? prev[s - 1] : NEGV;
    float mx = fmaxf(a, sh);
    float ea = __expf(a - mx), eb = __expf(sh - mx);
    float inv = 1.0f / (ea + eb);
    out.x = ea * inv; out.y = eb * inv;
  }
  pi[rowp + s] = out;
}

// ======================================================================
extern "C" void kernel_launch(void* const* d_in, const int* in_sizes, int n_in,
                              void* d_out, int out_size, void* d_ws, size_t ws_size,
                              hipStream_t stream) {
  (void)in_sizes; (void)n_in; (void)out_size; (void)ws_size;
  const float* inputs  = (const float*)d_in[0];
  const float* src_enc = (const float*)d_in[1];
  const float* mask    = (const float*)d_in[2];
  const float* w1 = (const float*)d_in[3];
  const float* b1 = (const float*)d_in[4];
  const float* w2 = (const float*)d_in[5];
  const float* b2 = (const float*)d_in[6];
  const float* w3 = (const float*)d_in[7];
  const float* b3 = (const float*)d_in[8];

  const int B = Bc, T = Tc, S = Sc, NM = NMc, H = Hc, K = Kc;
  const size_t BTS = (size_t)B * T * S;

  // d_out layout: mu | pi | alignments | Wspec
  float* mu      = (float*)d_out;
  float* pi      = mu + BTS;
  float* aligns  = mu + 3 * BTS;
  float* wspec_f = mu + 4 * BTS;

  // workspace layout (256B aligned regions)
  auto up = [](size_t x) { return (x + 255) & ~(size_t)255; };
  char* w = (char*)d_ws;
  size_t off = 0;
  __bf16* x_bf  = (__bf16*)(w + off); off += up((size_t)B * T * NM * 2);
  __bf16* h1_bf = (__bf16*)(w + off); off += up((size_t)B * T * H * 2);  // reused as wspec_bf
  __bf16* h2_bf = (__bf16*)(w + off); off += up((size_t)B * T * H * 2);
  __bf16* s_bf  = (__bf16*)(w + off); off += up((size_t)B * S * H * 2);
  __bf16* w1t   = (__bf16*)(w + off); off += up((size_t)K * NM * H * 2); // [H, K*NM]
  __bf16* w2t   = (__bf16*)(w + off); off += up((size_t)K * H * H * 2);  // [H, K*H]
  __bf16* w3t   = (__bf16*)(w + off); off += up((size_t)K * H * H * 2);
  float*  part  = (float*)(w + off);  off += up((size_t)NZ * B * S * 2 * 4);
  float*  stats = (float*)(w + off);  off += up((size_t)B * S * 2 * 4);

  auto gsz = [](long long n) {
    long long g = (n + 255) / 256;
    return (int)(g > 4096 ? 4096 : g);
  };

  // 1) fp32 -> bf16 staging (+ weight transpose to [Cout, K*Cin])
  cvt_f32_bf16<<<gsz((long long)B*T*NM), 256, 0, stream>>>(inputs,  x_bf, (long long)B*T*NM);
  cvt_f32_bf16<<<gsz((long long)B*S*H),  256, 0, stream>>>(src_enc, s_bf, (long long)B*S*H);
  cvt_w_transpose<<<(K*NM*H + 255)/256, 256, 0, stream>>>(w1, w1t, K*NM, H);
  cvt_w_transpose<<<(K*H*H  + 255)/256, 256, 0, stream>>>(w2, w2t, K*H,  H);
  cvt_w_transpose<<<(K*H*H  + 255)/256, 256, 0, stream>>>(w3, w3t, K*H,  H);

  // 2) PostNet: three WMMA convs with tanh
  dim3 cg(T / 16, H / 64, B);
  conv_tanh_wmma<NMc><<<cg, 128, 0, stream>>>(x_bf,  w1t, b1, h1_bf, nullptr, T, H);
  conv_tanh_wmma<Hc> <<<cg, 128, 0, stream>>>(h1_bf, w2t, b2, h2_bf, nullptr, T, H);
  conv_tanh_wmma<Hc> <<<cg, 128, 0, stream>>>(h2_bf, w3t, b3, h1_bf, wspec_f, T, H);

  // 3) attention scores -> alignments slice of d_out
  int stiles = (S + 15) / 16;
  dim3 eg(T / 16, (stiles + 3) / 4, B);
  einsum_wmma<<<eg, 128, 0, stream>>>(h1_bf, s_bf, aligns);

  // 4) softmax over T (in place in d_out), T-chunked for parallelism
  dim3 sg((S + 255) / 256, B, NZ);
  colstats_partial<<<sg, 256, 0, stream>>>(aligns, part, T, S, B);
  colstats_merge<<<(B * S + 255) / 256, 256, 0, stream>>>(part, stats, S, B);
  softmax_norm<<<sg, 256, 0, stream>>>(aligns, stats, T, S, B);

  // 5) DP: serial mu scan (lean), then fully-parallel pi
  dp_mu_scan<<<B, 512, 0, stream>>>(aligns, mask, mu, T, S);
  dp_pi<<<dim3(T, B), 512, 0, stream>>>(mu, (float2*)pi, T, S);
}
